// MoELayer_18408229831102
// MI455X (gfx1250) — matrix operand order repaired and checked by reading, not compiled, not run on hardware
//
#include <hip/hip_runtime.h>
#include <hip/hip_bf16.h>

typedef __bf16 bf16_t;
typedef __attribute__((ext_vector_type(8)))  __bf16 v8bf;
typedef __attribute__((ext_vector_type(16))) __bf16 v16bf;
typedef __attribute__((ext_vector_type(8)))  float  v8f;

#define BM 256
#define BN 64
#define BK 32
#define LDST 40   // padded LDS row stride (bf16 elems); 80B rows, 16B aligned chunks

__device__ __forceinline__ v8bf cvt8(const float4 a, const float4 b) {
    v8bf r;
    r[0]=(bf16_t)a.x; r[1]=(bf16_t)a.y; r[2]=(bf16_t)a.z; r[3]=(bf16_t)a.w;
    r[4]=(bf16_t)b.x; r[5]=(bf16_t)b.y; r[6]=(bf16_t)b.z; r[7]=(bf16_t)b.w;
    return r;
}

__device__ __forceinline__ v16bf mk16(v8bf lo, v8bf hi) {
    return __builtin_shufflevector(lo, hi,
        0,1,2,3,4,5,6,7,8,9,10,11,12,13,14,15);
}

// C[row, col] = act( sum_k A[row,k] * W[col,k] + bias[col] )
// A = [A0 | A1] along k (K0 then K1 columns); lda1==0 broadcasts one row of A1.
// ACT: 0 = none, 1 = relu, 2 = sigmoid
// Requires: Ntok % 256 == 0, Ncols % 64 == 0, K0 % 32 == 0, (K0+K1) % 32 == 0.
template <int ACT>
__global__ __launch_bounds__(256)
void gemm_wmma(const float* __restrict__ A0, int lda0, int K0,
               const float* __restrict__ A1, int lda1, int K1,
               const float* __restrict__ W,  const float* __restrict__ bias,
               float* __restrict__ C, int Ncols)
{
    __shared__ bf16_t Alds[2][BM][LDST];
    __shared__ bf16_t Wlds[2][BN][LDST];

    const int tid    = threadIdx.x;
    const int wave   = tid >> 5;
    const int lane   = tid & 31;
    const int lanelo = lane & 15;
    const int lanehi = lane >> 4;
    const int Ktot   = K0 + K1;
    const int blockM = blockIdx.y * BM;
    const int blockN = blockIdx.x * BN;

    v8f acc[8] = {};

    // cooperative-load indices: one 32-float row of A per thread, 8 W floats per thread
    const int arow  = tid;             // 0..255
    const int wrow  = tid >> 2;        // 0..63
    const int wcolb = (tid & 3) * 8;   // 0,8,16,24

    auto stage = [&](int buf, int kk) {
        // ---- A tile row (fp32 -> bf16); kk and K0 are multiples of 32 ----
        const float* src = (kk < K0)
            ? A0 + (size_t)(blockM + arow) * lda0 + kk
            : A1 + (size_t)(blockM + arow) * lda1 + (kk - K0);
        float4 f[8];
#pragma unroll
        for (int i = 0; i < 8; ++i) f[i] = *(const float4*)(src + 4 * i);
#pragma unroll
        for (int i = 0; i < 4; ++i)
            *(v8bf*)&Alds[buf][arow][8 * i] = cvt8(f[2 * i], f[2 * i + 1]);
        // ---- W tile (fp32 -> bf16) ----
        const float* wsrc = W + (size_t)(blockN + wrow) * Ktot + kk + wcolb;
        float4 w0 = *(const float4*)(wsrc + 0);
        float4 w1 = *(const float4*)(wsrc + 4);
        *(v8bf*)&Wlds[buf][wrow][wcolb] = cvt8(w0, w1);
    };

    stage(0, 0);
    __syncthreads();

    const int nsteps = Ktot / BK;
    for (int s = 0; s < nsteps; ++s) {
        const int cur = s & 1;
        if (s + 1 < nsteps) stage(cur ^ 1, (s + 1) * BK);   // prefetch next tile

        // ---- load all fragments, then 8 back-to-back WMMAs ----
        const int m0 = 16 * wave + lanelo;                  // first M-tile row
        const int m1 = m0 + 128;                            // second M-tile row
        const int kb = lanehi * 8;                          // ISA 16-bit A layout
        v16bf a0 = mk16(*(const v8bf*)&Alds[cur][m0][kb],
                        *(const v8bf*)&Alds[cur][m0][kb + 16]);
        v16bf a1 = mk16(*(const v8bf*)&Alds[cur][m1][kb],
                        *(const v8bf*)&Alds[cur][m1][kb + 16]);
        const int kb2 = lanehi * 8;                         // ISA B layout
        v16bf bfr[4];
#pragma unroll
        for (int j = 0; j < 4; ++j) {
            const int n = 16 * j + lanelo;
            bfr[j] = mk16(*(const v8bf*)&Wlds[cur][n][kb2],
                          *(const v8bf*)&Wlds[cur][n][kb2 + 16]);
        }
#pragma unroll
        for (int j = 0; j < 4; ++j) {
            acc[j]     = __builtin_amdgcn_wmma_f32_16x16x32_bf16(
                false, a0, false, bfr[j], (short)0, acc[j],     false, false);
            acc[4 + j] = __builtin_amdgcn_wmma_f32_16x16x32_bf16(
                false, a1, false, bfr[j], (short)0, acc[4 + j], false, false);
        }
        __syncthreads();
    }

    // ---- epilogue: bias + activation + store (C/D layout: row = 8*lanehi + r) ----
#pragma unroll
    for (int j = 0; j < 4; ++j) {
        const int col = blockN + 16 * j + lanelo;
        const float bv = bias[col];
#pragma unroll
        for (int r = 0; r < 8; ++r) {
            const int row0 = blockM + 16 * wave + 8 * lanehi + r;
            float v0 = acc[j][r] + bv;
            float v1 = acc[4 + j][r] + bv;
            if (ACT == 1) { v0 = fmaxf(v0, 0.0f); v1 = fmaxf(v1, 0.0f); }
            else if (ACT == 2) {
                v0 = 1.0f / (1.0f + __expf(-v0));
                v1 = 1.0f / (1.0f + __expf(-v1));
            }
            C[(size_t)row0 * Ncols + col]         = v0;
            C[(size_t)(row0 + 128) * Ncols + col] = v1;
        }
    }
}

// ---- router: logits + attr softmax + top-2 mask + entropy, one block per token ----
__global__ __launch_bounds__(256)
void router_probs_kernel(const float* __restrict__ h,    // [N,1024]
                         const float* __restrict__ te,   // [N,4,64]
                         const float* __restrict__ r_w,  // [8,1024]
                         const float* __restrict__ r_b,  // [8]
                         const float* __restrict__ r_attr, // [64,8]
                         float* __restrict__ probs,      // [N,8]
                         float* __restrict__ entp)       // [N]
{
    __shared__ float shp[8][32];
    __shared__ float shsc[4][8];
    __shared__ float shlog[8];
    const int n = blockIdx.x;
    const int t = threadIdx.x;
    const int e = t & 7, s = t >> 3;   // 8 experts x 32 partials

    float part = 0.0f;
    for (int k = s; k < 1024; k += 32)
        part += h[(size_t)n * 1024 + k] * r_w[e * 1024 + k];
    shp[e][s] = part;

    if (t < 32) {
        const int task = t >> 3, ee = t & 7;
        float sc = 0.0f;
        for (int d = 0; d < 64; ++d)
            sc += te[((size_t)n * 4 + task) * 64 + d] * r_attr[d * 8 + ee];
        shsc[task][ee] = sc;
    }
    __syncthreads();
    if (t < 8) {
        float sum = 0.0f;
        for (int s2 = 0; s2 < 32; ++s2) sum += shp[t][s2];
        shlog[t] = sum + r_b[t];
    }
    __syncthreads();
    if (t == 0) {
        float p[8];
        float mx = shlog[0];
        for (int i = 1; i < 8; ++i) mx = fmaxf(mx, shlog[i]);
        float den = 0.0f;
        for (int i = 0; i < 8; ++i) { p[i] = __expf(shlog[i] - mx); den += p[i]; }
        for (int i = 0; i < 8; ++i) p[i] /= den;

        float ap[8] = {};
        for (int task = 0; task < 4; ++task) {
            float m2 = shsc[task][0];
            for (int i = 1; i < 8; ++i) m2 = fmaxf(m2, shsc[task][i]);
            float d2 = 0.0f, q[8];
            for (int i = 0; i < 8; ++i) { q[i] = __expf(shsc[task][i] - m2); d2 += q[i]; }
            for (int i = 0; i < 8; ++i) ap[i] += q[i] / d2;
        }
        for (int i = 0; i < 8; ++i) p[i] *= ap[i] * 0.25f;

        int i1 = 0;
        for (int i = 1; i < 8; ++i) if (p[i] > p[i1]) i1 = i;
        int i2 = -1;
        for (int i = 0; i < 8; ++i) { if (i == i1) continue; if (i2 < 0 || p[i] > p[i2]) i2 = i; }

        float ent = 0.0f;
        for (int i = 0; i < 8; ++i) {
            float v = (i == i1 || i == i2) ? p[i] : 0.0f;
            probs[(size_t)n * 8 + i] = v;
            ent += v * __logf(v + 1e-8f);
        }
        entp[n] = ent;
    }
}

__global__ __launch_bounds__(256)
void ent_reduce_kernel(const float* __restrict__ entp, float* __restrict__ out_scalar)
{
    __shared__ float s[256];
    const int t = threadIdx.x;
    float sum = 0.0f;
    for (int i = t; i < 4096; i += 256) sum += entp[i];
    s[t] = sum; __syncthreads();
    for (int off = 128; off > 0; off >>= 1) {
        if (t < off) s[t] += s[t + off];
        __syncthreads();
    }
    if (t == 0) *out_scalar = -s[0] * (1.0f / 4096.0f);
}

// ap[e,h] = attr_emb[e] . attr_proj_w[e,h,:] + attr_proj_b[e,h]
__global__ __launch_bounds__(256)
void attr_proj_kernel(const float* __restrict__ emb, const float* __restrict__ w,
                      const float* __restrict__ b, float* __restrict__ ap)
{
    const int gid = blockIdx.x * 256 + threadIdx.x;  // 8192
    const int e = gid >> 10, hh = gid & 1023;
    float s = 0.0f;
    for (int a = 0; a < 256; ++a)
        s += emb[e * 256 + a] * w[((size_t)e * 1024 + hh) * 256 + a];
    ap[gid] = s + b[gid];
}

// h = x*gate + ap*(1-gate)
__global__ __launch_bounds__(256)
void mix_kernel(const float* __restrict__ x, const float* __restrict__ gate,
                const float* __restrict__ ap_e, float* __restrict__ hbuf)
{
    const int gid = blockIdx.x * 256 + threadIdx.x;
    const int hh = gid & 1023;
    const float g = gate[gid];
    hbuf[gid] = x[gid] * g + ap_e[hh] * (1.0f - g);
}

// out[n,:] += probs[n,e] * LayerNorm(h2[n,:] + x[n,:]) ; one block per token
__global__ __launch_bounds__(256)
void ln_accum_kernel(const float* __restrict__ h2, const float* __restrict__ x,
                     const float* __restrict__ g, const float* __restrict__ b,
                     const float* __restrict__ probs, int e, float* __restrict__ out)
{
    __shared__ float s1[256], s2[256];
    const int n = blockIdx.x, t = threadIdx.x;
    float v[4]; float lsum = 0.0f, lsq = 0.0f;
#pragma unroll
    for (int i = 0; i < 4; ++i) {
        const int hh = t + 256 * i;
        const float val = h2[(size_t)n * 1024 + hh] + x[(size_t)n * 1024 + hh];
        v[i] = val; lsum += val; lsq += val * val;
    }
    s1[t] = lsum; s2[t] = lsq; __syncthreads();
    for (int off = 128; off > 0; off >>= 1) {
        if (t < off) { s1[t] += s1[t + off]; s2[t] += s2[t + off]; }
        __syncthreads();
    }
    const float mu  = s1[0] * (1.0f / 1024.0f);
    const float var = s2[0] * (1.0f / 1024.0f) - mu * mu;
    const float inv = rsqrtf(var + 1e-5f);
    const float p   = probs[(size_t)n * 8 + e];
#pragma unroll
    for (int i = 0; i < 4; ++i) {
        const int hh = t + 256 * i;
        const float y = (v[i] - mu) * inv * g[e * 1024 + hh] + b[e * 1024 + hh];
        out[(size_t)n * 1024 + hh] += p * y;
    }
}

extern "C" void kernel_launch(void* const* d_in, const int* in_sizes, int n_in,
                              void* d_out, int out_size, void* d_ws, size_t ws_size,
                              hipStream_t stream) {
    (void)in_sizes; (void)n_in; (void)ws_size;
    const float* x      = (const float*)d_in[0];
    const float* te     = (const float*)d_in[1];
    const float* a_emb  = (const float*)d_in[2];
    const float* apw    = (const float*)d_in[3];
    const float* apb    = (const float*)d_in[4];
    const float* gw1    = (const float*)d_in[5];
    const float* gb1    = (const float*)d_in[6];
    const float* gw2    = (const float*)d_in[7];
    const float* gb2    = (const float*)d_in[8];
    const float* fw1    = (const float*)d_in[9];
    const float* fb1    = (const float*)d_in[10];
    const float* fw2    = (const float*)d_in[11];
    const float* fb2    = (const float*)d_in[12];
    const float* lng    = (const float*)d_in[13];
    const float* lnb    = (const float*)d_in[14];
    const float* r_attr = (const float*)d_in[15];
    const float* r_in_w = (const float*)d_in[16];
    const float* r_in_b = (const float*)d_in[17];
    const float* r_mid_w= (const float*)d_in[18];
    const float* r_mid_b= (const float*)d_in[19];
    const float* r_w    = (const float*)d_in[20];
    const float* r_b    = (const float*)d_in[21];

    float* out = (float*)d_out;
    float* ws  = (float*)d_ws;
    const size_t N = 4096;     // B*S
    float* regA  = ws;                       // N*4096 (router h1 / expert I-dim acts)
    float* regB  = regA + N * 4096;          // N*1024
    float* regC  = regB + N * 1024;          // N*1024
    float* probs = regC + N * 1024;          // N*8
    float* entp  = probs + N * 8;            // N
    float* ap    = entp + N;                 // 8*1024

    hipMemsetAsync(d_out, 0, (size_t)out_size * sizeof(float), stream);

    const dim3 blk(256);
    // Router: h1 = relu([x|te] @ r_in_w^T + b)  [N,4096]
    gemm_wmma<1><<<dim3(4096 / BN, (int)N / BM), blk, 0, stream>>>(
        x, 1024, 1024, te, 256, 256, r_in_w, r_in_b, regA, 4096);
    // h2 = relu(h1 @ r_mid_w^T + b)  [N,1024]
    gemm_wmma<1><<<dim3(1024 / BN, (int)N / BM), blk, 0, stream>>>(
        regA, 4096, 4096, nullptr, 0, 0, r_mid_w, r_mid_b, regB, 1024);
    router_probs_kernel<<<(int)N, 256, 0, stream>>>(regB, te, r_w, r_b, r_attr, probs, entp);
    ent_reduce_kernel<<<1, 256, 0, stream>>>(entp, out + (size_t)4096 * 1024);
    attr_proj_kernel<<<32, 256, 0, stream>>>(a_emb, apw, apb, ap);

    for (int e = 0; e < 8; ++e) {
        const float* gw1e = gw1 + (size_t)e * 2048 * 2048;
        const float* gb1e = gb1 + (size_t)e * 2048;
        const float* gw2e = gw2 + (size_t)e * 1024 * 2048;
        const float* gb2e = gb2 + (size_t)e * 1024;
        const float* fw1e = fw1 + (size_t)e * 2048 * 1024;
        const float* fb1e = fb1 + (size_t)e * 2048;
        const float* fw2e = fw2 + (size_t)e * 1024 * 2048;
        const float* fb2e = fb2 + (size_t)e * 1024;
        const float* ap_e = ap + (size_t)e * 1024;

        // g1 = relu([x | bcast(ap_e)] @ gw1^T + b)  [N,2048]
        gemm_wmma<1><<<dim3(2048 / BN, (int)N / BM), blk, 0, stream>>>(
            x, 1024, 1024, ap_e, 0, 1024, gw1e, gb1e, regA, 2048);
        // gate = sigmoid(g1 @ gw2^T + b)  [N,1024]
        gemm_wmma<2><<<dim3(1024 / BN, (int)N / BM), blk, 0, stream>>>(
            regA, 2048, 2048, nullptr, 0, 0, gw2e, gb2e, regB, 1024);
        // h = x*gate + ap*(1-gate)
        mix_kernel<<<(int)(N * 1024 / 256), 256, 0, stream>>>(x, regB, ap_e, regC);
        // f1 = relu(h @ fw1^T + b)  [N,2048]
        gemm_wmma<1><<<dim3(2048 / BN, (int)N / BM), blk, 0, stream>>>(
            regC, 1024, 1024, nullptr, 0, 0, fw1e, fb1e, regA, 2048);
        // h2 = f1 @ fw2^T + b  [N,1024]
        gemm_wmma<0><<<dim3(1024 / BN, (int)N / BM), blk, 0, stream>>>(
            regA, 2048, 2048, nullptr, 0, 0, fw2e, fb2e, regB, 1024);
        // out += probs[:,e] * LN(h2 + x)
        ln_accum_kernel<<<(int)N, 256, 0, stream>>>(regB, x, lng, lnb, probs, e, out);
    }
}
